// WSAD_2654289789384
// MI455X (gfx1250) — matrix operand order repaired and checked by reading, not compiled, not run on hardware
//
#include <hip/hip_runtime.h>
#include <math.h>

// CDNA5 / gfx1250 fused WSAD kernel, round 3.
// Roofline: x = 268 MB fp32, single pass @23.3 TB/s ~= 11.5us floor.
// GEMM: V_WMMA_F32_16X16X32_BF16 with bf16 hi/lo error compensation
// (xh*mh + xh*ml + xl*mh, ~2^-17 rel err). Mem banks pre-split to bf16.
// Round-3 deltas: packed v2f stat accumulators (v_pk_fma_f32 instead of
// 96 scalar v_fmac per kstep) and a b128-vectorized top-4 scan, so the
// VALU pipe no longer outruns the matrix pipe.

typedef __attribute__((ext_vector_type(2)))  float  v2f;
typedef __attribute__((ext_vector_type(4)))  float  v4f;
typedef __attribute__((ext_vector_type(8)))  float  v8f;
typedef __attribute__((ext_vector_type(16))) float  v16f;
typedef __attribute__((ext_vector_type(16))) __bf16 v16bf;

#define DIM      512
#define KSTEPS   16       // DIM / 32
#define MEMSZ    60
#define TLEN     4096
#define WAVES_PB 4        // 128 threads / block

typedef union { v16f v; v8f h[2]; v2f p[8]; } U16;

static __device__ __forceinline__ void split_bf16(const v16f x, v16bf& hi, v16bf& lo) {
    hi = __builtin_convertvector(x, v16bf);
    v16f hf = __builtin_convertvector(hi, v16f);
    v16f r  = x - hf;
    lo = __builtin_convertvector(r, v16bf);
}

// ---- pre-kernel: split both mem banks into bf16 hi/lo, 128 slots x 512 ----
// slots 0..59 = nor_mem, 60..63 = 0-pad, 64..123 = abn_mem, 124..127 = 0-pad.
__global__ __launch_bounds__(256)
void pack_mem_bf16(const float* __restrict__ nmem,
                   const float* __restrict__ amem,
                   __bf16* __restrict__ mhi,
                   __bf16* __restrict__ mlo)
{
    int tid  = blockIdx.x * 256 + threadIdx.x;   // 0 .. 128*512-1
    int slot = tid >> 9;                         // 0..127
    int c    = tid & 511;
    int bank = slot >> 6;
    int s    = slot & 63;
    float v  = 0.f;
    if (s < MEMSZ) v = (bank ? amem : nmem)[s * DIM + c];
    __bf16 h = (__bf16)v;
    __bf16 l = (__bf16)(v - (float)h);
    mhi[tid] = h;
    mlo[tid] = l;
}

__global__ __launch_bounds__(WAVES_PB * 32)
void wsad_fused_kernel(const float* __restrict__ x,
                       const __bf16* __restrict__ mhi,
                       const __bf16* __restrict__ mlo,
                       const float* __restrict__ clsw,
                       const float* __restrict__ clsb,
                       float* __restrict__ out,
                       long BT)
{
    // 132-float pitch: rows 4 banks apart -> conflict-free stores and scans,
    // and 16B alignment for the v4f scan reads.
    __shared__ float attLds[WAVES_PB][16][132];

    const int lane = threadIdx.x & 31;
    const int w    = threadIdx.x >> 5;
    const int g    = lane >> 4;       // lane group (K-range select)
    const int m    = lane & 15;       // A row / B,C column

    const long tile = (long)blockIdx.x * WAVES_PB + w;
    const long row  = tile * 16 + m;
    const int  t    = (int)(row & (TLEN - 1));
    const long rowL = (t == 0)        ? row : row - 1;
    const long rowR = (t == TLEN - 1) ? row : row + 1;

    // A-fragment bases: lane covers cols {32ks+16g+0..15}
    const float* ax = x + row  * DIM + 16 * g;
    const float* lx = x + rowL * DIM + 16 * g;
    const float* rx = x + rowR * DIM + 16 * g;
    const float* wx = clsw + 16 * g;

    // B-fragment base: lane(g,n=m) holds mem[slot][32ks+16g .. +15], one 32B
    // v16bf load. Global slot for N-tile j is j*16+m (banks concatenated).
    const __bf16* ph = mhi + (long)m * DIM + 16 * g;
    const __bf16* pl = mlo + (long)m * DIM + 16 * g;

    v8f acc[8];
#pragma unroll
    for (int j = 0; j < 8; ++j)
        acc[j] = (v8f){0.f, 0.f, 0.f, 0.f, 0.f, 0.f, 0.f, 0.f};

    // packed (pairwise) stat accumulators -> v_pk_fma_f32
    v2f sq2 = {0.f, 0.f}, nl2 = {0.f, 0.f}, nr2 = {0.f, 0.f};
    v2f dl2 = {0.f, 0.f}, dr2 = {0.f, 0.f}, cw2 = {0.f, 0.f};

#pragma unroll 2
    for (int ks = 0; ks < KSTEPS; ++ks) {
        const int off = ks * 32;
        U16 A, XL, XR, WV;
        A.h[0]  = *(const v8f*)(ax + off);
        A.h[1]  = *(const v8f*)(ax + off + 8);
        XL.h[0] = *(const v8f*)(lx + off);
        XL.h[1] = *(const v8f*)(lx + off + 8);
        XR.h[0] = *(const v8f*)(rx + off);
        XR.h[1] = *(const v8f*)(rx + off + 8);
        WV.h[0] = *(const v8f*)(wx + off);
        WV.h[1] = *(const v8f*)(wx + off + 8);

        v16bf ahi, alo;
        split_bf16(A.v, ahi, alo);

#pragma unroll
        for (int j = 0; j < 8; ++j) {
            const long bo = (long)j * 16 * DIM + off;
            v16bf bh = *(const v16bf*)(ph + bo);
            v16bf bl = *(const v16bf*)(pl + bo);
            acc[j] = __builtin_amdgcn_wmma_f32_16x16x32_bf16(
                false, ahi, false, bh, (short)0, acc[j], false, false);
            acc[j] = __builtin_amdgcn_wmma_f32_16x16x32_bf16(
                false, ahi, false, bl, (short)0, acc[j], false, false);
            acc[j] = __builtin_amdgcn_wmma_f32_16x16x32_bf16(
                false, alo, false, bh, (short)0, acc[j], false, false);
        }

#pragma unroll
        for (int p = 0; p < 8; ++p) {
            v2f av = A.p[p], lv = XL.p[p], rv = XR.p[p], wv = WV.p[p];
            sq2 += av * av;
            nl2 += lv * lv;
            nr2 += rv * rv;
            dl2 += av * lv;
            dr2 += av * rv;
            cw2 += av * wv;
        }
    }

    float sumsq = sq2.x + sq2.y;
    float nsqL  = nl2.x + nl2.y;
    float nsqR  = nr2.x + nr2.y;
    float dotL  = dl2.x + dl2.y;
    float dotR  = dr2.x + dr2.y;
    float cls   = cw2.x + cw2.y;

    // lane m / m+16 cover complementary column sets of row m: fold.
    sumsq += __shfl_xor(sumsq, 16, 32);
    nsqL  += __shfl_xor(nsqL, 16, 32);
    nsqR  += __shfl_xor(nsqR, 16, 32);
    dotL  += __shfl_xor(dotL, 16, 32);
    dotR  += __shfl_xor(dotR, 16, 32);
    cls   += __shfl_xor(cls, 16, 32);

    // C layout: lane group g, VGPR v -> M = v + 8g, N = m.
#pragma unroll
    for (int j = 0; j < 8; ++j)
#pragma unroll
        for (int v = 0; v < 8; ++v)
            attLds[w][v + 8 * g][j * 16 + m] = acc[j][v];

    __syncthreads();

    // group 0 scans nor bank (cols 0..59), group 1 scans abn (cols 64..123);
    // pads never scanned. v4f reads + branchless online top-4 on raw dots
    // (sigmoid and 1/sqrt(D) are monotonic -> select first, transform after).
    const float* rowp = &attLds[w][m][g * 64];
    float t0 = -1e30f, t1 = -1e30f, t2 = -1e30f, t3 = -1e30f;
#pragma unroll
    for (int s4 = 0; s4 < MEMSZ / 4; ++s4) {
        v4f q = *(const v4f*)(rowp + 4 * s4);
#pragma unroll
        for (int e = 0; e < 4; ++e) {
            float v  = q[e];
            float u0 = fmaxf(t0, v);  float c1 = fminf(t0, v);
            float u1 = fmaxf(t1, c1); float c2 = fminf(t1, c1);
            float u2 = fmaxf(t2, c2); float c3 = fminf(t2, c2);
            float u3 = fmaxf(t3, c3);
            t0 = u0; t1 = u1; t2 = u2; t3 = u3;
        }
    }
    const float sc = 0.044194173824159216f;   // 1/sqrt(512)
    float e0 = 1.f / (1.f + __expf(-t0 * sc));
    float e1 = 1.f / (1.f + __expf(-t1 * sc));
    float e2 = 1.f / (1.f + __expf(-t2 * sc));
    float e3 = 1.f / (1.f + __expf(-t3 * sc));
    float attMean   = 0.25f * (e0 + e1 + e2 + e3);
    float otherMean = __shfl_xor(attMean, 16, 32);
    float N_att = g ? otherMean : attMean;
    float A_att = g ? attMean   : otherMean;

    float norm = sqrtf(sumsq);
    float cosL = dotL / fmaxf(norm * sqrtf(nsqL), 1e-8f);
    float cosR = dotR / fmaxf(norm * sqrtf(nsqR), 1e-8f);
    float cosScore = (2.f - (cosL + cosR)) * 0.25f;

    float preAtt   = 1.f / (1.f + __expf(-(cls + clsb[0])));
    float memScore = (A_att + 1.f - N_att) * 0.5f;
    float finalS   = (memScore + preAtt) * 0.5f;

    if (g == 0 && row < BT) {
        out[row]          = finalS;
        out[BT + row]     = preAtt;
        out[2 * BT + row] = cosScore;
    }
}

extern "C" void kernel_launch(void* const* d_in, const int* in_sizes, int n_in,
                              void* d_out, int out_size, void* d_ws, size_t ws_size,
                              hipStream_t stream) {
    const float* x    = (const float*)d_in[0];
    const float* nmem = (const float*)d_in[1];
    const float* amem = (const float*)d_in[2];
    const float* clsw = (const float*)d_in[3];
    const float* clsb = (const float*)d_in[4];
    float* out = (float*)d_out;

    // Workspace: bf16 hi + lo copies of the concatenated mem banks.
    const long MEMELEMS = 128L * DIM;                 // 65536
    __bf16* mhi = (__bf16*)d_ws;
    __bf16* mlo = mhi + MEMELEMS;                     // 128 KB offset

    pack_mem_bf16<<<(int)(MEMELEMS / 256), 256, 0, stream>>>(nmem, amem, mhi, mlo);

    const long BT     = (long)in_sizes[0] / DIM;      // 131072 rows
    const long tiles  = BT / 16;                      // 8192 wave tiles
    const int  blocks = (int)((tiles + WAVES_PB - 1) / WAVES_PB);

    wsad_fused_kernel<<<blocks, WAVES_PB * 32, 0, stream>>>(
        x, mhi, mlo, clsw, clsb, out, BT);
}
